// Decompressor_274877907636
// MI455X (gfx1250) — compile-verified, matrix operand order checked by reading, not run on hardware
//
#include <hip/hip_runtime.h>
#include <hip/hip_bf16.h>

#define B_SZ    256
#define H_SZ    1024
#define H4      4096
#define D_OUT   128
#define T_STEPS 128

// GEMM blocking: block = 128x128 (8 waves, each 32x64). K staged in BK=32 chunks.
#define BM 128
#define BN 128
#define BK 32
#define STR 40                       // LDS row stride (bf16): 80B, 16B-aligned, bank-clean
#define ABUF_ELEMS (BM * STR)        // 5120 bf16
#define BBUF_ELEMS (BN * STR)        // 5120 bf16
#define BUF_ELEMS  (ABUF_ELEMS + BBUF_ELEMS)   // 10240 bf16 = 20480 B per stage
#define NBUF 3                       // triple buffer: 60 KB LDS

typedef __attribute__((ext_vector_type(16))) __bf16 bf16x16;
typedef __attribute__((ext_vector_type(8)))  __bf16 bf16x8;
typedef __attribute__((ext_vector_type(8)))  float  f32x8;

__device__ __forceinline__ __bf16 f2bf(float f) {
    unsigned u = __builtin_bit_cast(unsigned, f);
    unsigned r = u + 0x7FFFu + ((u >> 16) & 1u);   // round-to-nearest-even
    unsigned short h = (unsigned short)(r >> 16);
    return __builtin_bit_cast(__bf16, h);
}

__device__ __forceinline__ float sigmoidf(float x) {
    return 1.0f / (1.0f + __expf(-x));
}

// 16x32 bf16 fragment from an LDS-staged chunk (row-major, stride STR).
// Lane m=lane%16 -> row row_base+m; half=lane/16 -> K chunks at half*8, half*8+16.
__device__ __forceinline__ bf16x16 load_frag_lds(const __bf16* s, int row_base,
                                                 int m, int half) {
    const __bf16* q = s + (size_t)(row_base + m) * STR + half * 8;
    bf16x8 lo = *reinterpret_cast<const bf16x8*>(q);
    bf16x8 hi = *reinterpret_cast<const bf16x8*>(q + 16);
    bf16x16 r;
#pragma unroll
    for (int i = 0; i < 8; ++i) { r[i] = lo[i]; r[i + 8] = hi[i]; }
    return r;
}

// Async-copy one 128-row x 32-k bf16 slab into LDS (2 b128 transfers / thread).
__device__ __forceinline__ void stage_slab(const __bf16* __restrict__ M,
                                           int first_row, int K, int kb,
                                           unsigned lds_byte_base, int tid) {
#pragma unroll
    for (int p = 0; p < 2; ++p) {
        int e   = tid + p * 256;        // 512 x 16B transfers = 128x32 bf16
        int row = e >> 2;
        int seg = e & 3;
        const __bf16* gp = M + (size_t)(first_row + row) * K + kb + seg * 8;
        unsigned lo = lds_byte_base + (unsigned)(row * (STR * 2) + seg * 16);
        asm volatile("global_load_async_to_lds_b128 %0, %1, off"
                     :: "v"(lo), "v"(gp) : "memory");
    }
}

// ---------------------------------------------------------------------------
// Prep: bf16 weight conversion, W_sum = W_ih + W_hh, b_sum, x->bf16, c = 0.
// ---------------------------------------------------------------------------
__global__ __launch_bounds__(256)
void prep_kernel(const float* __restrict__ x,
                 const float* __restrict__ W_ih, const float* __restrict__ W_hh,
                 const float* __restrict__ b_ih, const float* __restrict__ b_hh,
                 const float* __restrict__ W_out,
                 __bf16* __restrict__ Wih_b, __bf16* __restrict__ Wsum_b,
                 __bf16* __restrict__ Wout_b, float* __restrict__ b_sum,
                 __bf16* __restrict__ x_b, float* __restrict__ c_state) {
    size_t tid    = (size_t)blockIdx.x * blockDim.x + threadIdx.x;
    size_t stride = (size_t)gridDim.x * blockDim.x;
    const size_t total = (size_t)H4 * H_SZ;
    for (size_t i = tid; i < total; i += stride) {
        float wi = W_ih[i], wh = W_hh[i];
        Wih_b[i]  = f2bf(wi);
        Wsum_b[i] = f2bf(wi + wh);
        if (i < (size_t)D_OUT * H_SZ) Wout_b[i] = f2bf(W_out[i]);
        if (i < (size_t)H4)           b_sum[i]  = b_ih[i] + b_hh[i];
        if (i < (size_t)B_SZ * H_SZ) { x_b[i] = f2bf(x[i]); c_state[i] = 0.0f; }
    }
}

// ---------------------------------------------------------------------------
// WMMA GEMM: C[m,n] = sum_k A[m,k]*W[n,k] + bias[n]
// A [256 x K], W [N x K], both bf16 row-major, both staged through LDS with
// triple-buffered async copies. Block 128x128; wave w: rows (w&3)*32..+31,
// cols (w>>2)*64..+63. Grid: (M/128, N/128).
// ---------------------------------------------------------------------------
__global__ __launch_bounds__(256)
void wmma_gemm_kernel(const __bf16* __restrict__ A, const __bf16* __restrict__ W,
                      const float* __restrict__ bias, float* __restrict__ out,
                      int K, int out_row_stride, int out_col_off) {
    __shared__ __align__(16) __bf16 smem[NBUF * BUF_ELEMS];   // 60 KB

    const int tid  = threadIdx.x;
    const int lane = tid & 31;
    const int m    = lane & 15;
    const int half = lane >> 4;
    const int w    = tid >> 5;          // 8 waves
    const int r    = w & 3;             // 32-row quad in block
    const int g    = w >> 2;            // 64-col group in block

    const int block_row = blockIdx.x * BM;
    const int block_n   = blockIdx.y * BN;
    const int n_base    = block_n + g * 64;
    const unsigned lds0      = (unsigned)(size_t)(&smem[0]);
    const unsigned buf_bytes = BUF_ELEMS * 2;
    const unsigned a_bytes   = ABUF_ELEMS * 2;
    const int NC = K / BK;              // 32 chunks for K=1024

    // Prologue: chunks 0,1,2 in flight (4 async instrs each per thread).
    stage_slab(A, block_row, K, 0 * BK, lds0 + 0 * buf_bytes, tid);
    stage_slab(W, block_n,   K, 0 * BK, lds0 + 0 * buf_bytes + a_bytes, tid);
    stage_slab(A, block_row, K, 1 * BK, lds0 + 1 * buf_bytes, tid);
    stage_slab(W, block_n,   K, 1 * BK, lds0 + 1 * buf_bytes + a_bytes, tid);
    stage_slab(A, block_row, K, 2 * BK, lds0 + 2 * buf_bytes, tid);
    stage_slab(W, block_n,   K, 2 * BK, lds0 + 2 * buf_bytes + a_bytes, tid);

    f32x8 acc[2][4] = {{f32x8{0}, f32x8{0}, f32x8{0}, f32x8{0}},
                       {f32x8{0}, f32x8{0}, f32x8{0}, f32x8{0}}};

    for (int c = 0; c < NC; ++c) {
        // Outstanding asyncs: chunk c (4) + chunk c+1 (4) + chunk c+2 (4).
        const int rem = NC - 1 - c;
        if (rem >= 2)      asm volatile("s_wait_asynccnt 0x8" ::: "memory");
        else if (rem == 1) asm volatile("s_wait_asynccnt 0x4" ::: "memory");
        else               asm volatile("s_wait_asynccnt 0x0" ::: "memory");
        __syncthreads();   // chunk c visible block-wide

        const unsigned buf = (unsigned)(c % NBUF);
        const __bf16* sa = smem + (size_t)buf * BUF_ELEMS;
        const __bf16* sb = sa + ABUF_ELEMS;

        bf16x16 a0 = load_frag_lds(sa, r * 32,      m, half);
        bf16x16 a1 = load_frag_lds(sa, r * 32 + 16, m, half);
#pragma unroll
        for (int j = 0; j < 4; ++j) {
            bf16x16 b = load_frag_lds(sb, g * 64 + j * 16, m, half);
            acc[0][j] = __builtin_amdgcn_wmma_f32_16x16x32_bf16(
                false, a0, false, b, (short)0, acc[0][j], false, false);
            acc[1][j] = __builtin_amdgcn_wmma_f32_16x16x32_bf16(
                false, a1, false, b, (short)0, acc[1][j], false, false);
        }

        __syncthreads();   // all waves done reading buffer c%3
        if (c + 3 < NC) {
            const unsigned dst = lds0 + buf * buf_bytes;
            stage_slab(A, block_row, K, (c + 3) * BK, dst,           tid);
            stage_slab(W, block_n,   K, (c + 3) * BK, dst + a_bytes, tid);
        }
    }

    // D layout: lane holds column n_base + lane%16; VGPR i -> row half*8 + i.
#pragma unroll
    for (int s = 0; s < 2; ++s) {
        const int r0 = block_row + r * 32 + s * 16 + half * 8;
#pragma unroll
        for (int j = 0; j < 4; ++j) {
            const int col = n_base + j * 16 + m;
            const float bv = bias[col];
#pragma unroll
            for (int i = 0; i < 8; ++i) {
                out[(size_t)(r0 + i) * out_row_stride + out_col_off + col] =
                    acc[s][j][i] + bv;
            }
        }
    }
}

// ---------------------------------------------------------------------------
// LSTM cell elementwise: gates [B x 4H] (i,f,g,o blocks) -> c (f32), h (bf16)
// ---------------------------------------------------------------------------
__global__ __launch_bounds__(256)
void lstm_cell_kernel(const float* __restrict__ gates,
                      float* __restrict__ c_state, __bf16* __restrict__ h_b) {
    int idx = blockIdx.x * blockDim.x + threadIdx.x;
    if (idx >= B_SZ * H_SZ) return;
    int b = idx >> 10;           // / H
    int j = idx & (H_SZ - 1);    // % H
    const float* gp = gates + (size_t)b * H4;
    float i_g = gp[j];
    float f_g = gp[H_SZ + j];
    float g_g = gp[2 * H_SZ + j];
    float o_g = gp[3 * H_SZ + j];
    float c_new = sigmoidf(f_g) * c_state[idx] + sigmoidf(i_g) * tanhf(g_g);
    float h_new = sigmoidf(o_g) * tanhf(c_new);
    c_state[idx] = c_new;
    h_b[idx]     = f2bf(h_new);
}

// ---------------------------------------------------------------------------
extern "C" void kernel_launch(void* const* d_in, const int* in_sizes, int n_in,
                              void* d_out, int out_size, void* d_ws, size_t ws_size,
                              hipStream_t stream) {
    const float* x     = (const float*)d_in[0];
    const float* W_ih  = (const float*)d_in[1];
    const float* W_hh  = (const float*)d_in[2];
    const float* b_ih  = (const float*)d_in[3];
    const float* b_hh  = (const float*)d_in[4];
    const float* W_out = (const float*)d_in[5];
    const float* b_out = (const float*)d_in[6];
    float* out = (float*)d_out;

    // Workspace carve-up (256B aligned slabs)
    char* ws = (char*)d_ws;
    size_t off = 0;
    auto alloc = [&](size_t bytes) {
        void* p = ws + off;
        off += (bytes + 255) & ~(size_t)255;
        return p;
    };
    __bf16* Wih_b  = (__bf16*)alloc((size_t)H4 * H_SZ * 2);     // 8 MiB
    __bf16* Wsum_b = (__bf16*)alloc((size_t)H4 * H_SZ * 2);     // 8 MiB
    __bf16* Wout_b = (__bf16*)alloc((size_t)D_OUT * H_SZ * 2);  // 256 KiB
    float*  b_sum  = (float*) alloc((size_t)H4 * 4);            // 16 KiB
    __bf16* x_b    = (__bf16*)alloc((size_t)B_SZ * H_SZ * 2);   // 512 KiB
    float*  gates  = (float*) alloc((size_t)B_SZ * H4 * 4);     // 4 MiB
    float*  c_st   = (float*) alloc((size_t)B_SZ * H_SZ * 4);   // 1 MiB
    __bf16* h_b    = (__bf16*)alloc((size_t)B_SZ * H_SZ * 2);   // 512 KiB

    prep_kernel<<<2048, 256, 0, stream>>>(x, W_ih, W_hh, b_ih, b_hh, W_out,
                                          Wih_b, Wsum_b, Wout_b, b_sum, x_b, c_st);

    for (int t = 0; t < T_STEPS; ++t) {
        const __bf16* A = (t == 0) ? x_b   : h_b;
        const __bf16* W = (t == 0) ? Wih_b : Wsum_b;

        // gates = A @ W^T + b_sum : grid (256/128, 4096/128)
        wmma_gemm_kernel<<<dim3(2, 32), 256, 0, stream>>>(
            A, W, b_sum, gates, H_SZ, H4, 0);
        // elementwise cell update
        lstm_cell_kernel<<<(B_SZ * H_SZ) / 256, 256, 0, stream>>>(gates, c_st, h_b);
        // out[:, t, :] = h_t @ W_out^T + b_out : grid (256/128, 128/128)
        wmma_gemm_kernel<<<dim3(2, 1), 256, 0, stream>>>(
            h_b, Wout_b, b_out, out, H_SZ, T_STEPS * D_OUT, t * D_OUT);
    }
}
